// Autoencoder_10874857193743
// MI455X (gfx1250) — matrix operand order repaired and checked by reading, not compiled, not run on hardware
//
#include <hip/hip_runtime.h>
#include <math.h>

// ---------------- problem constants ----------------
#define BB   16
#define TT   512
#define SS   320
#define HH   512
#define KC   1024          // codes per codebook
#define NQ   4
#define BT   (BB*TT)       // 8192
#define G4   (4*HH)        // 2048

typedef __attribute__((ext_vector_type(16))) __bf16 v16bf;
typedef __attribute__((ext_vector_type(8)))  float  v8f;

// ---------------- helpers ----------------
__device__ __forceinline__ __bf16 f2bf(float f) {
  union { float f; unsigned u; } x; x.f = f;
  unsigned r = x.u + 0x7FFFu + ((x.u >> 16) & 1u);   // round-to-nearest-even
  unsigned short s = (unsigned short)(r >> 16);
  __bf16 b; __builtin_memcpy(&b, &s, 2); return b;
}

__device__ __forceinline__ float sigm(float x) { return 1.f / (1.f + __expf(-x)); }

// A fragment: 16x32 bf16, row-major (M x K) source.
// lane half h: row = m0 + (lane&15), K in {k0+8h..+7} U {k0+16+8h..+7}
__device__ __forceinline__ v16bf load_a_frag(const __bf16* A, int ld, int m0, int k0, int lane) {
  int r  = m0 + (lane & 15);
  int kb = (lane & 16) ? 8 : 0;
  const __bf16* p = A + (size_t)r * ld + k0 + kb;
  union { v16bf v; uint4 u[2]; } f;
  f.u[0] = *(const uint4*)p;
  f.u[1] = *(const uint4*)(p + 16);
  return f.v;
}

// B fragment: 32x16 bf16 from (N,K)-major storage (i.e. B^T rows).
// lane half h: col = n0 + (lane&15), K in {k0+8h..+7} U {k0+16+8h..+7}
__device__ __forceinline__ v16bf load_b_frag(const __bf16* Bt, int ld, int n0, int k0, int lane) {
  int n  = n0 + (lane & 15);
  int kb = (lane & 16) ? 8 : 0;
  const __bf16* p = Bt + (size_t)n * ld + k0 + kb;
  union { v16bf v; uint4 u[2]; } f;
  f.u[0] = *(const uint4*)p;
  f.u[1] = *(const uint4*)(p + 16);
  return f.v;
}

__device__ __forceinline__ v8f wmma_bf16(v16bf a, v16bf b, v8f c) {
  return __builtin_amdgcn_wmma_f32_16x16x32_bf16(false, a, false, b, (short)0, c, false, false);
}

// grid-wide barrier (two-counter, generation based)
__device__ __forceinline__ void grid_barrier(unsigned* bar, int nwg) {
  __syncthreads();
  if (threadIdx.x == 0) {
    unsigned g = atomicAdd(&bar[1], 0u);
    __threadfence();
    unsigned my = atomicAdd(&bar[0], 1u);
    if (my == (unsigned)nwg - 1u) {
      atomicExch(&bar[0], 0u);
      __threadfence();
      atomicAdd(&bar[1], 1u);
    } else {
      while (atomicAdd(&bar[1], 0u) == g) __builtin_amdgcn_s_sleep(1);
    }
  }
  __syncthreads();
}

// ---------------- small prep kernels ----------------
__global__ void k_init(float* quant, unsigned* bar, float* loss) {
  size_t i = (size_t)blockIdx.x * 256 + threadIdx.x;
  if (i < (size_t)BT * HH) quant[i] = 0.f;
  if (i < 4) loss[i] = 0.f;
  if (i < 8) bar[i] = 0u;
}

__global__ void k_f2bf(__bf16* dst, const float* src, int n) {
  int i = blockIdx.x * 256 + threadIdx.x;
  if (i < n) dst[i] = f2bf(src[i]);
}

// dst[(n)*K + k] = bf16(src[k*N + n])   (make (N,K)-major bf16 copy)
__global__ void k_tr_bf(__bf16* dst, const float* src, int K, int N) {
  int i = blockIdx.x * 256 + threadIdx.x;
  if (i < K * N) {
    int k = i / N, n = i - k * N;
    dst[(size_t)n * K + k] = f2bf(src[i]);
  }
}

__global__ void k_cbnorm(float* cbn, const float* cb) {   // 0.5*||E||^2, NQ*KC codes
  int c = blockIdx.x * 256 + threadIdx.x;
  if (c < NQ * KC) {
    const float* p = cb + (size_t)c * HH;
    float s = 0.f;
    for (int h = 0; h < HH; ++h) s += p[h] * p[h];
    cbn[c] = 0.5f * s;
  }
}

__global__ void k_addb(float* bs, const float* a, const float* b, int n) {
  int i = blockIdx.x * 256 + threadIdx.x;
  if (i < n) bs[i] = a[i] + b[i];
}

// ---------------- generic bf16 WMMA GEMM: C = A(MxK) * Bt(NxK)^T + bias ----------------
// block = 256 threads = 8 waves laid out 4(M) x 2(N); each wave owns a 32x32 tile
// -> block tile 128 x 64.  4 WMMAs per K-chunk off 2 A-frags + 2 B-frags (2x reuse).
__global__ void k_gemm(const __bf16* __restrict__ A, const __bf16* __restrict__ Bt,
                       const float* __restrict__ bias, float* __restrict__ C,
                       int M, int N, int K) {
  int wave = threadIdx.x >> 5, lane = threadIdx.x & 31;
  int m0 = blockIdx.x * 128 + (wave & 3) * 32;
  int n0 = blockIdx.y * 64 + (wave >> 2) * 32;
  v8f acc00 = {}, acc01 = {}, acc10 = {}, acc11 = {};
  for (int k0 = 0; k0 < K; k0 += 32) {
    v16bf a0 = load_a_frag(A, K, m0,      k0, lane);
    v16bf a1 = load_a_frag(A, K, m0 + 16, k0, lane);
    v16bf b0 = load_b_frag(Bt, K, n0,      k0, lane);
    v16bf b1 = load_b_frag(Bt, K, n0 + 16, k0, lane);
    acc00 = wmma_bf16(a0, b0, acc00);
    acc01 = wmma_bf16(a0, b1, acc01);
    acc10 = wmma_bf16(a1, b0, acc10);
    acc11 = wmma_bf16(a1, b1, acc11);
  }
  int half = lane >> 4, cl = lane & 15;
  int c0 = n0 + cl, c1 = n0 + 16 + cl;
  float bv0 = bias ? bias[c0] : 0.f;
  float bv1 = bias ? bias[c1] : 0.f;
#pragma unroll
  for (int i = 0; i < 8; ++i) {
    size_t r0 = (size_t)(m0 + i + 8 * half) * N;
    size_t r1 = (size_t)(m0 + 16 + i + 8 * half) * N;
    C[r0 + c0] = acc00[i] + bv0;
    C[r0 + c1] = acc01[i] + bv1;
    C[r1 + c0] = acc10[i] + bv0;
    C[r1 + c1] = acc11[i] + bv1;
  }
}

// ---------------- LayerNorm + ReLU -> residual (f32 + bf16) ----------------
__global__ void k_ln_relu(const float* __restrict__ x, const float* __restrict__ g,
                          const float* __restrict__ bta, float* __restrict__ r,
                          __bf16* __restrict__ rbf) {
  __shared__ float red[256];
  int m = blockIdx.x, tid = threadIdx.x;
  size_t base = (size_t)m * HH;
  float v0 = x[base + tid], v1 = x[base + tid + 256];
  red[tid] = v0 + v1; __syncthreads();
  for (int o = 128; o > 0; o >>= 1) { if (tid < o) red[tid] += red[tid + o]; __syncthreads(); }
  float mu = red[0] * (1.f / (float)HH);
  __syncthreads();
  float d0 = v0 - mu, d1 = v1 - mu;
  red[tid] = d0 * d0 + d1 * d1; __syncthreads();
  for (int o = 128; o > 0; o >>= 1) { if (tid < o) red[tid] += red[tid + o]; __syncthreads(); }
  float rstd = rsqrtf(red[0] * (1.f / (float)HH) + 1e-5f);
  float o0 = fmaxf(0.f, d0 * rstd * g[tid] + bta[tid]);
  float o1 = fmaxf(0.f, d1 * rstd * g[tid + 256] + bta[tid + 256]);
  r[base + tid] = o0;        rbf[base + tid] = f2bf(o0);
  r[base + tid + 256] = o1;  rbf[base + tid + 256] = f2bf(o1);
}

// ---------------- VQ: fused score GEMM + running argmax ----------------
// block: 8 waves x 16 rows = 128 rows; the wave's A operand (16 rows x K=512,
// 16 fragments = 128 VGPRs) is register-resident and reused for all 64 code
// tiles, so the hot loop issues only B-fragment loads (1 wmma / 2 b128 loads).
__global__ void k_vq_dist(const __bf16* __restrict__ rbf, const __bf16* __restrict__ cb,
                          const float* __restrict__ cbn, int* __restrict__ idxOut) {
  __shared__ float sv[8][16][16];
  __shared__ int   si[8][16][16];
  int tid = threadIdx.x, wave = tid >> 5, lane = tid & 31;
  int m0 = blockIdx.x * 128 + wave * 16;
  int half = lane >> 4, cloc = lane & 15;

  v16bf af[HH / 32];
#pragma unroll
  for (int kc = 0; kc < HH / 32; ++kc) af[kc] = load_a_frag(rbf, HH, m0, kc * 32, lane);

  float best[8]; int bidx[8];
#pragma unroll
  for (int i = 0; i < 8; ++i) { best[i] = -3.4e38f; bidx[i] = 0; }
  for (int ct = 0; ct < KC / 16; ++ct) {
    v8f acc = {};
#pragma unroll
    for (int kc = 0; kc < HH / 32; ++kc) {
      v16bf b = load_b_frag(cb, HH, ct * 16, kc * 32, lane);
      acc = wmma_bf16(af[kc], b, acc);
    }
    int code = ct * 16 + cloc;
    float cn = cbn[code];
#pragma unroll
    for (int i = 0; i < 8; ++i) {
      float s = acc[i] - cn;                       // argmin d == argmax (r.E - 0.5||E||^2)
      if (s > best[i]) { best[i] = s; bidx[i] = code; }
    }
  }
#pragma unroll
  for (int i = 0; i < 8; ++i) { sv[wave][i + 8 * half][cloc] = best[i]; si[wave][i + 8 * half][cloc] = bidx[i]; }
  __syncthreads();
  if (lane < 16) {
    float bv = sv[wave][lane][0]; int bid = si[wave][lane][0];
    for (int c = 1; c < 16; ++c) {
      float v = sv[wave][lane][c]; int id = si[wave][lane][c];
      if (v > bv || (v == bv && id < bid)) { bv = v; bid = id; }
    }
    idxOut[m0 + lane] = bid;
  }
}

__global__ void k_vq_update(const float* __restrict__ cbq, const int* __restrict__ idx,
                            float* __restrict__ res, __bf16* __restrict__ rbf,
                            float* __restrict__ quant, __bf16* __restrict__ qbf,
                            float* __restrict__ lossOut, int last) {
  __shared__ float red[256];
  int m = blockIdx.x, tid = threadIdx.x;
  const float* e = cbq + (size_t)idx[m] * HH;
  float ls = 0.f;
  for (int h = tid; h < HH; h += 256) {
    size_t o = (size_t)m * HH + h;
    float r = res[o], qv = e[h];
    float d = qv - r; ls += d * d;
    float nq = quant[o] + qv; quant[o] = nq;
    float nr = r - qv; res[o] = nr; rbf[o] = f2bf(nr);
    if (last) qbf[o] = f2bf(nq);
  }
  red[tid] = ls; __syncthreads();
  for (int o = 128; o > 0; o >>= 1) { if (tid < o) red[tid] += red[tid + o]; __syncthreads(); }
  if (tid == 0) atomicAdd(lossOut, red[0] * (1.f / ((float)BT * (float)HH)));
}

// ---------------- LSTM recurrence: persistent, 16 WGs, 1 grid-barrier / step ----------------
// WG w owns hidden slice j in [32w, 32w+32): computes its i/f/g/o gate columns
// ({g*512 + j} for g=0..3) with bf16 WMMA against h(t), keeps its c slice in LDS.
// Each wave's Wh^T slice (16 fragments = 128 VGPRs) is loaded ONCE and kept
// register-resident for all 512 timesteps; the per-step loop only reloads h(t).
__global__ void k_lstm(const float* __restrict__ pre, const __bf16* __restrict__ whbf,
                       float* __restrict__ y, __bf16* __restrict__ ybf,
                       __bf16* __restrict__ hbuf, unsigned* __restrict__ bar) {
  __shared__ float gvals[4][BB][32];
  __shared__ float cst[BB][32];
  const int tid = threadIdx.x;
  const int wave = tid >> 5, lane = tid & 31;
  const int nwg = gridDim.x;
  const int j0 = blockIdx.x * 32;
  const int gate = wave >> 1;
  const int col0 = gate * HH + j0 + (wave & 1) * 16;
  const int half = lane >> 4, cloc = lane & 15;

  // weights resident in VGPRs for the whole recurrence
  v16bf bw[HH / 32];
#pragma unroll
  for (int kc = 0; kc < HH / 32; ++kc) bw[kc] = load_b_frag(whbf, HH, col0, kc * 32, lane);

  // zero our c slice and our slice of h(0) (buffer 0)
  for (int p = tid; p < BB * 32; p += 256) {
    int b = p >> 5, j = p & 31;
    cst[b][j] = 0.f;
    hbuf[(size_t)b * HH + j0 + j] = f2bf(0.f);
  }
  __threadfence();
  grid_barrier(bar, nwg);

  for (int t = 0; t < TT; ++t) {
    const __bf16* hb = hbuf + (size_t)(t & 1) * (BB * HH);
    v8f acc = {};
#pragma unroll
    for (int i = 0; i < 8; ++i) {
      int b = i + 8 * half;
      acc[i] = pre[((size_t)b * TT + t) * G4 + col0 + cloc];
    }
    if (t < TT - 1)
      __builtin_prefetch(&pre[((size_t)(8 * half) * TT + t + 1) * G4 + col0 + cloc], 0, 1);
#pragma unroll
    for (int kc = 0; kc < HH / 32; ++kc) {
      v16bf a = load_a_frag(hb, HH, 0, kc * 32, lane);       // A = h(t): 16 batches x 512
      acc = wmma_bf16(a, bw[kc], acc);
    }
    int jl = (wave & 1) * 16 + cloc;
#pragma unroll
    for (int i = 0; i < 8; ++i) gvals[gate][i + 8 * half][jl] = acc[i];
    __syncthreads();

    __bf16* hn = hbuf + (size_t)((t + 1) & 1) * (BB * HH);
    for (int p = tid; p < BB * 32; p += 256) {
      int b = p >> 5, j = p & 31;
      float ig = gvals[0][b][j], fg = gvals[1][b][j];
      float gg = gvals[2][b][j], og = gvals[3][b][j];
      float c = sigm(fg) * cst[b][j] + sigm(ig) * tanhf(gg);
      cst[b][j] = c;
      float h = sigm(og) * tanhf(c);
      size_t m = (size_t)b * TT + t;
      y[m * HH + j0 + j] = h;
      __bf16 hv = f2bf(h);
      ybf[m * HH + j0 + j] = hv;
      hn[(size_t)b * HH + j0 + j] = hv;
    }
    __threadfence();
    grid_barrier(bar, nwg);
  }
}

// ---------------- host launcher ----------------
extern "C" void kernel_launch(void* const* d_in, const int* in_sizes, int n_in,
                              void* d_out, int out_size, void* d_ws, size_t ws_size,
                              hipStream_t stream) {
  (void)in_sizes; (void)n_in; (void)out_size; (void)ws_size;
  const float* wave  = (const float*)d_in[0];
  const float* enc_w = (const float*)d_in[1];
  const float* enc_b = (const float*)d_in[2];
  const float* ln_g  = (const float*)d_in[3];
  const float* ln_b  = (const float*)d_in[4];
  const float* cbk   = (const float*)d_in[5];
  const float* wi    = (const float*)d_in[6];
  const float* wh    = (const float*)d_in[7];
  const float* bi    = (const float*)d_in[8];
  const float* bh    = (const float*)d_in[9];
  const float* dec_w = (const float*)d_in[10];
  const float* dec_b = (const float*)d_in[11];
  float* out = (float*)d_out;
  float* lossOut = out + (size_t)BT * SS;   // recon first, then 4 losses

  // workspace carve-up (256B aligned)
  char* ws = (char*)d_ws;
  size_t off = 0;
  auto carve = [&](size_t bytes) { size_t o = off; off = (off + bytes + 255) & ~(size_t)255; return (void*)(ws + o); };
  __bf16* wavebf = (__bf16*)carve((size_t)BT * SS * 2);
  __bf16* enc_wT = (__bf16*)carve((size_t)HH * SS * 2);
  __bf16* dec_wT = (__bf16*)carve((size_t)SS * HH * 2);
  __bf16* cb_bf  = (__bf16*)carve((size_t)NQ * KC * HH * 2);
  float*  cbn    = (float*)carve((size_t)NQ * KC * 4);
  __bf16* wi_bf  = (__bf16*)carve((size_t)2 * G4 * HH * 2);
  __bf16* wh_bf  = (__bf16*)carve((size_t)2 * G4 * HH * 2);
  float*  bsum   = (float*)carve((size_t)2 * G4 * 4);
  float*  res    = (float*)carve((size_t)BT * HH * 4);
  __bf16* rbf    = (__bf16*)carve((size_t)BT * HH * 2);
  float*  quant  = (float*)carve((size_t)BT * HH * 4);
  __bf16* qbf    = (__bf16*)carve((size_t)BT * HH * 2);
  int*    idxb   = (int*)carve((size_t)BT * 4);
  float*  pre    = (float*)carve((size_t)BT * G4 * 4);   // also used as encoder GEMM temp
  float*  ybuf   = (float*)carve((size_t)BT * HH * 4);
  __bf16* ybf    = (__bf16*)carve((size_t)BT * HH * 2);
  __bf16* hbuf   = (__bf16*)carve((size_t)2 * BB * HH * 2);
  unsigned* bar  = (unsigned*)carve(256);

  // ---- init + bf16 weight prep ----
  k_init<<<(BT * HH + 255) / 256, 256, 0, stream>>>(quant, bar, lossOut);
  k_f2bf<<<(BT * SS + 255) / 256, 256, 0, stream>>>(wavebf, wave, BT * SS);
  k_tr_bf<<<(SS * HH + 255) / 256, 256, 0, stream>>>(enc_wT, enc_w, SS, HH);  // (S,H)->(H,S)
  k_tr_bf<<<(HH * SS + 255) / 256, 256, 0, stream>>>(dec_wT, dec_w, HH, SS);  // (H,S)->(S,H)
  k_f2bf<<<(NQ * KC * HH + 255) / 256, 256, 0, stream>>>(cb_bf, cbk, NQ * KC * HH);
  k_cbnorm<<<(NQ * KC + 255) / 256, 256, 0, stream>>>(cbn, cbk);
  k_f2bf<<<(2 * G4 * HH + 255) / 256, 256, 0, stream>>>(wi_bf, wi, 2 * G4 * HH);
  k_f2bf<<<(2 * G4 * HH + 255) / 256, 256, 0, stream>>>(wh_bf, wh, 2 * G4 * HH);
  k_addb<<<(2 * G4 + 255) / 256, 256, 0, stream>>>(bsum, bi, bh, 2 * G4);

  // ---- encoder: frames @ enc_w + b -> pre(temp), then LN+ReLU -> residual ----
  k_gemm<<<dim3(BT / 128, HH / 64), 256, 0, stream>>>(wavebf, enc_wT, enc_b, pre, BT, HH, SS);
  k_ln_relu<<<BT, 256, 0, stream>>>(pre, ln_g, ln_b, res, rbf);

  // ---- residual VQ: 4 stages ----
  for (int q = 0; q < NQ; ++q) {
    k_vq_dist<<<BT / 128, 256, 0, stream>>>(rbf, cb_bf + (size_t)q * KC * HH,
                                            cbn + (size_t)q * KC, idxb);
    k_vq_update<<<BT, 256, 0, stream>>>(cbk + (size_t)q * KC * HH, idxb, res, rbf,
                                        quant, qbf, lossOut + q, q == NQ - 1 ? 1 : 0);
  }

  // ---- LSTM layer 0 ----
  k_gemm<<<dim3(BT / 128, G4 / 64), 256, 0, stream>>>(qbf, wi_bf, bsum, pre, BT, G4, HH);
  k_lstm<<<16, 256, 0, stream>>>(pre, wh_bf, ybuf, ybf, hbuf, bar);
  // ---- LSTM layer 1 ----
  k_gemm<<<dim3(BT / 128, G4 / 64), 256, 0, stream>>>(ybf, wi_bf + (size_t)G4 * HH,
                                                      bsum + G4, pre, BT, G4, HH);
  k_lstm<<<16, 256, 0, stream>>>(pre, wh_bf + (size_t)G4 * HH, ybuf, ybf, hbuf, bar);

  // ---- output projection -> recon ----
  k_gemm<<<dim3(BT / 128, SS / 64), 256, 0, stream>>>(ybf, dec_wT, dec_b, out, BT, SS, HH);
}
